// Attention_63660005261999
// MI455X (gfx1250) — matrix operand order repaired and checked by reading, not compiled, not run on hardware
//
#include <hip/hip_runtime.h>
#include <hip/hip_bf16.h>

// Problem constants (match reference setup_inputs)
#define QLEN 2048
#define MLEN 2048
#define KLEN 4096   // MLEN + QLEN
#define BSZ  8
#define DMODEL 1024
#define NEG_INF (-1000000.0f)

typedef __attribute__((ext_vector_type(8)))  float  v8f;
typedef __attribute__((ext_vector_type(16))) __bf16 v16bf;

union Frag { v16bf v; uint4 q[2]; };

// pack 4 floats -> 4 bf16, store as 8 bytes
__device__ __forceinline__ void st_bf4(__bf16* dst, float4 v) {
    union { __bf16 b[4]; uint2 u; } p;
    p.b[0] = (__bf16)v.x; p.b[1] = (__bf16)v.y;
    p.b[2] = (__bf16)v.z; p.b[3] = (__bf16)v.w;
    *(uint2*)dst = p.u;
}

__device__ __forceinline__ const float* c_row(const float* __restrict__ h,
                                              const float* __restrict__ mems,
                                              int kg, int b) {
    // c = concat(mems, h) along k; layout [k, b, d]
    return (kg < MLEN) ? (mems + ((size_t)kg * BSZ + b) * DMODEL)
                       : (h    + ((size_t)(kg - MLEN) * BSZ + b) * DMODEL);
}

// ---------------------------------------------------------------------------
// Kernel 1: S[b,q,k] = sum_d Q[q,b,d]*C[k,b,d]  (bf16 WMMA, f32 acc) + mask
// Tile: 128(q) x 128(k), d-chunks of 32, double-buffered LDS staging.
// 256 threads = 8 waves, 2x4 16x16 fragments per wave.
// ---------------------------------------------------------------------------
__global__ __launch_bounds__(256) void attn_score_kernel(
    const float* __restrict__ h, const float* __restrict__ mems,
    float* __restrict__ S) {
    const int kt = blockIdx.x, qt = blockIdx.y, b = blockIdx.z;
    const int q0 = qt * 128, k0 = kt * 128;
    const int t = threadIdx.x;

    // fully masked tile: k_min > q_max + mlen
    if (k0 > q0 + 127 + MLEN) {
        for (int i = t; i < 128 * 128; i += 256) {
            int r = i >> 7, c = i & 127;
            S[(size_t)(b * QLEN + q0 + r) * KLEN + (k0 + c)] = NEG_INF;
        }
        return;
    }

    __shared__ __bf16 qs[2][128 * 40];   // [buf][qrow*40 + d] bf16
    __shared__ __bf16 ks[2][128 * 40];   // [buf][krow*40 + d] bf16

    const int lane = t & 31, wave = t >> 5;
    const int mg = wave & 3;          // m-group: rows [mg*32, mg*32+32)
    const int ng = wave >> 2;         // n-group: cols [ng*64, ng*64+64)
    const int hrow = lane & 15, kh = lane >> 4;

    // per-thread staging slots: 4 float4 each for Q and K per chunk
    int sr[4], sc[4];
    const float* qptr[4];
    const float* kptr[4];
#pragma unroll
    for (int u = 0; u < 4; ++u) {
        int i = t + u * 256;
        sr[u] = i >> 3; sc[u] = (i & 7) * 4;
        qptr[u] = h + ((size_t)(q0 + sr[u]) * BSZ + b) * DMODEL + sc[u];
        kptr[u] = c_row(h, mems, k0 + sr[u], b) + sc[u];
    }

    v8f acc[2][4] = {};
    float4 vq[4], vk[4];

    // prologue: stage chunk 0 into buffer 0
#pragma unroll
    for (int u = 0; u < 4; ++u) { vq[u] = *(const float4*)(qptr[u]); vk[u] = *(const float4*)(kptr[u]); }
#pragma unroll
    for (int u = 0; u < 4; ++u) {
        st_bf4(&qs[0][sr[u] * 40 + sc[u]], vq[u]);
        st_bf4(&ks[0][sr[u] * 40 + sc[u]], vk[u]);
    }

    const int NITER = DMODEL / 32;
    for (int it = 0; it < NITER; ++it) {
        const int cur = it & 1;
        __syncthreads();                       // buf[cur] ready for everyone
        // prefetch next chunk from global into registers
        if (it + 1 < NITER) {
            const int dk = (it + 1) * 32;
#pragma unroll
            for (int u = 0; u < 4; ++u) {
                vq[u] = *(const float4*)(qptr[u] + dk);
                vk[u] = *(const float4*)(kptr[u] + dk);
            }
        }
        // load all fragments, then issue 8 WMMAs back-to-back
        const __bf16* qb = qs[cur];
        const __bf16* kb = ks[cur];
        Frag a[2], bf[4];
#pragma unroll
        for (int i = 0; i < 2; ++i) {
            const __bf16* p = &qb[((mg * 2 + i) * 16 + hrow) * 40 + kh * 8];
            a[i].q[0] = *(const uint4*)(p);
            a[i].q[1] = *(const uint4*)(p + 16);
        }
#pragma unroll
        for (int j = 0; j < 4; ++j) {
            const __bf16* p = &kb[((ng * 4 + j) * 16 + hrow) * 40 + kh * 8];
            bf[j].q[0] = *(const uint4*)(p);
            bf[j].q[1] = *(const uint4*)(p + 16);
        }
#pragma unroll
        for (int j = 0; j < 4; ++j)
#pragma unroll
            for (int i = 0; i < 2; ++i)
                acc[i][j] = __builtin_amdgcn_wmma_f32_16x16x32_bf16(
                    false, a[i].v, false, bf[j].v, (short)0, acc[i][j], false, false);
        // store prefetched chunk into the other buffer
        if (it + 1 < NITER) {
#pragma unroll
            for (int u = 0; u < 4; ++u) {
                st_bf4(&qs[cur ^ 1][sr[u] * 40 + sc[u]], vq[u]);
                st_bf4(&ks[cur ^ 1][sr[u] * 40 + sc[u]], vk[u]);
            }
        }
    }

    // store with causal mask (k > q + mlen -> NEG_INF)
#pragma unroll
    for (int i = 0; i < 2; ++i)
#pragma unroll
        for (int j = 0; j < 4; ++j)
#pragma unroll
            for (int r = 0; r < 8; ++r) {
                int ql = (mg * 2 + i) * 16 + kh * 8 + r;
                int kl = (ng * 4 + j) * 16 + hrow;
                int qg = q0 + ql, kg = k0 + kl;
                float v = (kg > qg + MLEN) ? NEG_INF : acc[i][j][r];
                S[(size_t)(b * QLEN + qg) * KLEN + kg] = v;
            }
}

// ---------------------------------------------------------------------------
// Kernel 2: rowwise softmax over k (4096 f32) -> bf16 P written in-place over
// the first half of the row's f32 storage (one block owns one row).
// ---------------------------------------------------------------------------
__global__ __launch_bounds__(256) void attn_softmax_kernel(float* __restrict__ ws) {
    const size_t row = blockIdx.x;
    float* x = ws + row * KLEN;
    const int t = threadIdx.x, lane = t & 31, wave = t >> 5;
    __shared__ float red[8];

    float vals[16];
    float m = -3.4e38f;
#pragma unroll
    for (int j = 0; j < 4; ++j) {
        float4 v = *(const float4*)(x + t * 16 + j * 4);
        vals[j * 4 + 0] = v.x; vals[j * 4 + 1] = v.y;
        vals[j * 4 + 2] = v.z; vals[j * 4 + 3] = v.w;
        m = fmaxf(m, fmaxf(fmaxf(v.x, v.y), fmaxf(v.z, v.w)));
    }
#pragma unroll
    for (int off = 16; off; off >>= 1) m = fmaxf(m, __shfl_xor(m, off, 32));
    if (lane == 0) red[wave] = m;
    __syncthreads();
    m = red[0];
#pragma unroll
    for (int w = 1; w < 8; ++w) m = fmaxf(m, red[w]);

    float e[16], s = 0.f;
#pragma unroll
    for (int i = 0; i < 16; ++i) { e[i] = expf(vals[i] - m); s += e[i]; }
#pragma unroll
    for (int off = 16; off; off >>= 1) s += __shfl_xor(s, off, 32);
    __syncthreads();
    if (lane == 0) red[wave] = s;
    __syncthreads();
    s = 0.f;
#pragma unroll
    for (int w = 0; w < 8; ++w) s += red[w];
    const float inv = 1.0f / s;

    __bf16* p = (__bf16*)x;   // safe: all reads of this row done before barrier
#pragma unroll
    for (int j = 0; j < 4; ++j) {
        float4 v = make_float4(e[j * 4 + 0] * inv, e[j * 4 + 1] * inv,
                               e[j * 4 + 2] * inv, e[j * 4 + 3] * inv);
        st_bf4(&p[t * 16 + j * 4], v);
    }
}

// ---------------------------------------------------------------------------
// Kernel 3: out[q,b,d] = sum_k P[b,q,k] * C[k,b,d]  (bf16 WMMA, f32 acc)
// Tile: 128(q) x 128(d), k-chunks of 32. P tile staged with CDNA5 async
// LDS loads (ASYNCcnt); C staged transposed (f32->bf16 conversion in VALU).
// ---------------------------------------------------------------------------
__global__ __launch_bounds__(256) void attn_out_kernel(
    const float* __restrict__ h, const float* __restrict__ mems,
    const void* __restrict__ ws, float* __restrict__ out) {
    const int nt = blockIdx.x, qt = blockIdx.y, b = blockIdx.z;
    const int q0 = qt * 128, n0 = nt * 128;
    const int t = threadIdx.x, lane = t & 31, wave = t >> 5;
    const int mg = wave & 3, ng = wave >> 2;
    const int hrow = lane & 15, kh = lane >> 4;

    __shared__ __bf16 ps[128 * 40];   // [qrow][kchunk] bf16
    __shared__ __bf16 ct[128 * 40];   // transposed: [dcol][krow] bf16

    v8f acc[2][4] = {};
    // P rows: bf16 over the first half of each 16 KB f32 row of the workspace
    const char* psad = (const char*)ws + (size_t)(b * QLEN + q0) * (KLEN * 4);
    const int kend = min(KLEN, q0 + 128 + MLEN);   // P == 0 beyond causal bound

    // per-thread async-copy slots for the P tile: 2 x 16B per thread
    unsigned prow[2], pseg[2], plds[2];
#pragma unroll
    for (int u = 0; u < 2; ++u) {
        int i = t + u * 256;
        prow[u] = i >> 2; pseg[u] = i & 3;
        plds[u] = (unsigned)(uintptr_t)&ps[prow[u] * 40 + pseg[u] * 8];
    }

    for (int kk = 0; kk < kend; kk += 32) {
        __syncthreads();
        // stage P tile via async global->LDS copies (tracked by ASYNCcnt)
#pragma unroll
        for (int u = 0; u < 2; ++u) {
            unsigned goff = prow[u] * (unsigned)(KLEN * 4) + (unsigned)kk * 2u + pseg[u] * 16u;
            asm volatile("global_load_async_to_lds_b128 %0, %1, %2"
                         :: "v"(plds[u]), "v"(goff), "s"(psad)
                         : "memory");
        }
        // stage C tile transposed: 32 k rows x 128 d cols -> ct[dcol][k]
        for (int i = t; i < 1024; i += 256) {
            int r = i >> 5, c = (i & 31) * 4;
            const float* cb = c_row(h, mems, kk + r, b);
            float4 v = *(const float4*)(cb + n0 + c);
            ct[(c + 0) * 40 + r] = (__bf16)v.x;
            ct[(c + 1) * 40 + r] = (__bf16)v.y;
            ct[(c + 2) * 40 + r] = (__bf16)v.z;
            ct[(c + 3) * 40 + r] = (__bf16)v.w;
        }
        asm volatile("s_wait_asynccnt 0x0" ::: "memory");
        __syncthreads();

        Frag a[2], bf[4];
#pragma unroll
        for (int i = 0; i < 2; ++i) {
            const __bf16* p = &ps[((mg * 2 + i) * 16 + hrow) * 40 + kh * 8];
            a[i].q[0] = *(const uint4*)(p);
            a[i].q[1] = *(const uint4*)(p + 16);
        }
#pragma unroll
        for (int j = 0; j < 4; ++j) {
            const __bf16* p = &ct[((ng * 4 + j) * 16 + hrow) * 40 + kh * 8];
            bf[j].q[0] = *(const uint4*)(p);
            bf[j].q[1] = *(const uint4*)(p + 16);
        }
#pragma unroll
        for (int j = 0; j < 4; ++j)
#pragma unroll
            for (int i = 0; i < 2; ++i)
                acc[i][j] = __builtin_amdgcn_wmma_f32_16x16x32_bf16(
                    false, a[i].v, false, bf[j].v, (short)0, acc[i][j], false, false);
    }

#pragma unroll
    for (int i = 0; i < 2; ++i)
#pragma unroll
        for (int j = 0; j < 4; ++j)
#pragma unroll
            for (int r = 0; r < 8; ++r) {
                int ql = (mg * 2 + i) * 16 + kh * 8 + r;
                int dl = (ng * 4 + j) * 16 + hrow;
                out[((size_t)(q0 + ql) * BSZ + b) * DMODEL + n0 + dl] = acc[i][j][r];
            }
}

// ---------------------------------------------------------------------------
extern "C" void kernel_launch(void* const* d_in, const int* in_sizes, int n_in,
                              void* d_out, int out_size, void* d_ws, size_t ws_size,
                              hipStream_t stream) {
    const float* h    = (const float*)d_in[0];  // [qlen, bsz, d]
    const float* mems = (const float*)d_in[1];  // [mlen, bsz, d]
    float* out = (float*)d_out;                 // [qlen, bsz, d]
    float* S   = (float*)d_ws;                  // needs BSZ*QLEN*KLEN*4 = 256 MB

    (void)in_sizes; (void)n_in; (void)out_size; (void)ws_size;

    // 1) scores (masked) -> ws as f32 [b, q, k]
    attn_score_kernel<<<dim3(KLEN / 128, QLEN / 128, BSZ), 256, 0, stream>>>(h, mems, S);
    // 2) rowwise softmax, write P as bf16 in-place (row stride stays 16 KB)
    attn_softmax_kernel<<<dim3(BSZ * QLEN), 256, 0, stream>>>(S);
    // 3) out = P @ C
    attn_out_kernel<<<dim3(DMODEL / 128, QLEN / 128, BSZ), 256, 0, stream>>>(h, mems, d_ws, out);
}
